// MultiLayerFusionAdapter_63273458204774
// MI455X (gfx1250) — compile-verified, matrix operand order checked by reading, not run on hardware
//
#include <hip/hip_runtime.h>
#include <hip/hip_bf16.h>
#include <stdint.h>

#define B_  4
#define S_  4096
#define A_  512
#define H_  1024
#define NH_ 16
#define HD_ 64

typedef __attribute__((ext_vector_type(16))) __bf16        v16bf;
typedef __attribute__((ext_vector_type(8)))  float         v8f;
typedef __attribute__((ext_vector_type(4)))  unsigned int  u32x4;

union FragU { v16bf f; u32x4 u[2]; };

// 16-half WMMA operand slice: halves 0..7 = K kb..kb+7, halves 8..15 = K kb+16..kb+23.
// Lane-half (lanes 0-15 vs 16-31) selects kb = k0 or k0+8 -> matches 16-bit 16x32 operand
// VGPR layout (ISA 7.12.2).
__device__ __forceinline__ v16bf load_frag16(const __bf16* p) {
  FragU r;
  r.u[0] = *(const u32x4*)(p);
  r.u[1] = *(const u32x4*)(p + 16);
  return r.f;
}

__device__ __forceinline__ v8f wmma_bf16(v16bf a, v16bf b, v8f c) {
  return __builtin_amdgcn_wmma_f32_16x16x32_bf16(false, a, false, b, (short)0, c, false, false);
}

// GLOBAL_LOAD_ASYNC_TO_LDS_B128 (GV mode): vdst = per-lane LDS byte offset, vaddr = 64-bit
// global address pair. Tracked with ASYNCcnt (ISA 10.x / 15.18.3 op 98).
__device__ __forceinline__ void async_load_b128(const __bf16* gptr, void* lptr) {
  const uint32_t loff = (uint32_t)(uintptr_t)lptr;   // generic LDS addr low bits = LDS offset
  asm volatile("global_load_async_to_lds_b128 %0, %1, off"
               :: "v"(loff), "v"((unsigned long long)(uintptr_t)gptr)
               : "memory");
}

#define WAIT_ASYNC(n) asm volatile("s_wait_asynccnt %0" :: "i"(n) : "memory")

__device__ __forceinline__ float sanf(float x) {
  if (x != x) return 0.0f;              // NaN -> 0
  x = fminf(x, 1.0e4f);                 // +inf -> 1e4
  x = fmaxf(x, -1.0e4f);                // -inf -> -1e4
  return x;
}

// ---------------- fp32 -> bf16 conversion (optionally sanitized) ----------------
__global__ void cvt_kernel(const float* __restrict__ src, __bf16* __restrict__ dst,
                           int n, int do_san) {
  int i = blockIdx.x * blockDim.x + threadIdx.x;
  if (i >= n) return;
  float x = src[i];
  if (do_san) x = sanf(x);
  dst[i] = (__bf16)x;
}

// ---------------- bf16 WMMA GEMM: out[M,N] = A[M,K] @ W[N,K]^T + bias ----------------
// Block = 8 waves = 128 rows x 64 cols of C. A- and W-tiles staged into LDS with
// double-buffered GLOBAL_LOAD_ASYNC_TO_LDS_B128; WMMA operands come from LDS.
// MODE 0: bf16 out [M,N]   MODE 1: bf16 out transposed [N,M]   MODE 2: fp32 out, san*rs
template<int MODE, int KC>
__global__ __launch_bounds__(256) void gemm_kernel(
    const __bf16* __restrict__ Aact, const __bf16* __restrict__ Wt,
    const float* __restrict__ bias,
    __bf16* __restrict__ outb, float* __restrict__ outf,
    const float* __restrict__ rs_ptr,
    int M, int N) {
  __shared__ __align__(16) __bf16 alds[2][128][32];
  __shared__ __align__(16) __bf16 wlds[2][64][32];

  const int tid  = threadIdx.x;
  const int wave = tid >> 5;
  const int lane = tid & 31;
  const int half = lane >> 4;
  const int idx  = lane & 15;
  const int m0   = blockIdx.x * 128;
  const int n0   = blockIdx.y * 64;

  // staging split: A tile 128x32 (2 x b128 per thread), W tile 64x32 (1 x b128)
  const int ar = tid >> 1;            // 0..127
  const int as = (tid & 1) * 16;      // 0 / 16
  const int wr = tid >> 2;            // 0..63
  const int ws = (tid & 3) * 8;       // 0,8,16,24

  const __bf16* abase = Aact + (size_t)(m0 + ar) * KC + as;
  const __bf16* wbase = Wt + (size_t)(n0 + wr) * KC + ws;
  auto stage = [&](int buf, int k0) {
    async_load_b128(abase + k0,     &alds[buf][ar][as]);
    async_load_b128(abase + k0 + 8, &alds[buf][ar][as + 8]);
    async_load_b128(wbase + k0,     &wlds[buf][wr][ws]);
  };

  const v8f vzero = {0.f,0.f,0.f,0.f,0.f,0.f,0.f,0.f};
  v8f acc[4];
  acc[0] = vzero; acc[1] = vzero; acc[2] = vzero; acc[3] = vzero;

  stage(0, 0);
  constexpr int KT = KC / 32;
#pragma unroll 2
  for (int kt = 0; kt < KT; ++kt) {
    const int buf = kt & 1;
    if (kt + 1 < KT) { stage(buf ^ 1, (kt + 1) * 32); WAIT_ASYNC(3); }
    else             { WAIT_ASYNC(0); }
    __syncthreads();                                   // stage kt visible block-wide

    // preload all fragments, then batch the WMMAs
    const v16bf af  = load_frag16(&alds[buf][wave * 16 + idx][half * 8]);
    v16bf wf[4];
#pragma unroll
    for (int t = 0; t < 4; ++t) wf[t] = load_frag16(&wlds[buf][t * 16 + idx][half * 8]);
#pragma unroll
    for (int t = 0; t < 4; ++t) acc[t] = wmma_bf16(af, wf[t], acc[t]);
    __syncthreads();                                   // done reading buf before re-stage
  }

  float rs = 1.0f;
  if (MODE == 2) {
    rs = rs_ptr[0];
    rs = fminf(fmaxf(rs, 0.0f), 0.3f);
  }
#pragma unroll
  for (int t = 0; t < 4; ++t) {
    const int n = n0 + t * 16 + idx;
    const float bv = bias[n];
#pragma unroll
    for (int v = 0; v < 8; ++v) {
      const int r = m0 + wave * 16 + v + half * 8;
      const float val = acc[t][v] + bv;
      if (MODE == 0) {
        outb[(size_t)r * N + n] = (__bf16)val;
      } else if (MODE == 1) {
        outb[(size_t)n * M + r] = (__bf16)val;
      } else {
        outf[(size_t)r * N + n] = sanf(val) * rs;
      }
    }
  }
}

// ---------------- flash attention ----------------
// 8 waves/block share (b, head): K/V 32-key chunks staged once per block into LDS with
// double-buffered async loads. Each wave owns a 16-query tile; online softmax in f32.
__global__ __launch_bounds__(256) void attn_kernel(
    const __bf16* __restrict__ Qb, const __bf16* __restrict__ Kb,
    const __bf16* __restrict__ Vt, const int* __restrict__ mask,
    __bf16* __restrict__ ctx) {
  __shared__ __align__(16) __bf16 klds[2][32][64];   // [key][dim of head]
  __shared__ __align__(16) __bf16 vlds[2][64][32];   // [dim of head][key]
  __shared__ __align__(16) __bf16 plds[8][16][32];   // per-wave probs tile

  const int tid  = threadIdx.x;
  const int wave = tid >> 5;
  const int lane = tid & 31;
  const int half = lane >> 4;
  const int idx  = lane & 15;

  const int tile    = blockIdx.x * 8 + wave;          // 8 consecutive s-tiles, same (b,h)
  const int s_tiles = S_ / 16;
  const int s0 = (tile % s_tiles) * 16;
  const int h  = (tile / s_tiles) % NH_;
  const int b  = tile / (s_tiles * NH_);

  const __bf16* qrow = Qb + ((size_t)b * S_ + s0 + idx) * H_ + h * HD_;
  const v16bf qf0 = load_frag16(qrow + half * 8);            // head dims 0..31
  const v16bf qf1 = load_frag16(qrow + 32 + half * 8);       // head dims 32..63

  // staging split: K chunk 32x64 (1 x b128/thread), V chunk 64x32 (1 x b128/thread)
  const int kr = tid >> 3, ks = (tid & 7) * 8;   // key row, dim seg
  const int vr = tid >> 2, vs = (tid & 3) * 8;   // dim row, key seg
  const __bf16* kbase = Kb + ((size_t)b * A_ + kr) * H_ + h * HD_ + ks;
  const __bf16* vbase = Vt + (size_t)(h * HD_ + vr) * (B_ * A_) + (size_t)b * A_ + vs;
  auto stage = [&](int buf, int a0) {
    async_load_b128(kbase + (size_t)a0 * H_, &klds[buf][kr][ks]);
    async_load_b128(vbase + a0,              &vlds[buf][vr][vs]);
  };

  const v8f vzero = {0.f,0.f,0.f,0.f,0.f,0.f,0.f,0.f};
  v8f acc[4];
  acc[0] = vzero; acc[1] = vzero; acc[2] = vzero; acc[3] = vzero;
  float mrow[8], lrow[8];
#pragma unroll
  for (int v = 0; v < 8; ++v) { mrow[v] = -1.0e30f; lrow[v] = 0.0f; }

  stage(0, 0);
#pragma unroll 2
  for (int a0 = 0; a0 < A_; a0 += 32) {
    const int buf = (a0 >> 5) & 1;
    if (a0 + 32 < A_) { stage(buf ^ 1, a0 + 32); WAIT_ASYNC(2); }
    else              { WAIT_ASYNC(0); }
    __syncthreads();                                   // chunk staged block-wide

    // preload all K fragments, then batch the score WMMAs
    const v16bf kf00 = load_frag16(&klds[buf][idx][half * 8]);
    const v16bf kf01 = load_frag16(&klds[buf][idx][32 + half * 8]);
    const v16bf kf10 = load_frag16(&klds[buf][16 + idx][half * 8]);
    const v16bf kf11 = load_frag16(&klds[buf][16 + idx][32 + half * 8]);
    v8f st0 = wmma_bf16(qf0, kf00, vzero);
    v8f st1 = wmma_bf16(qf0, kf10, vzero);
    st0 = wmma_bf16(qf1, kf01, st0);
    st1 = wmma_bf16(qf1, kf11, st1);

    const int mk0 = mask[b * A_ + a0 + idx];
    const int mk1 = mask[b * A_ + a0 + 16 + idx];

    float p0[8], p1[8];
#pragma unroll
    for (int v = 0; v < 8; ++v) {
      float x0 = fminf(fmaxf(st0[v] * 0.125f, -50.f), 50.f);
      float x1 = fminf(fmaxf(st1[v] * 0.125f, -50.f), 50.f);
      if (mk0 == 0) x0 = -50.f;      // clip(-10000, +-50) == -50
      if (mk1 == 0) x1 = -50.f;
      float cm = fmaxf(x0, x1);
#pragma unroll
      for (int off = 1; off < 16; off <<= 1)
        cm = fmaxf(cm, __shfl_xor(cm, off, 32));       // row max across key-lanes
      const float mn   = fmaxf(mrow[v], cm);
      const float corr = __expf(mrow[v] - mn);
      mrow[v] = mn;
      const float e0 = __expf(x0 - mn);
      const float e1 = __expf(x1 - mn);
      float ps = e0 + e1;
#pragma unroll
      for (int off = 1; off < 16; off <<= 1)
        ps += __shfl_xor(ps, off, 32);                 // row sum
      lrow[v] = lrow[v] * corr + ps;
#pragma unroll
      for (int t = 0; t < 4; ++t) acc[t][v] *= corr;
      p0[v] = e0; p1[v] = e1;
    }

    // re-layout probs (C layout -> A operand layout) through LDS
#pragma unroll
    for (int v = 0; v < 8; ++v) {
      const int r = v + half * 8;
      plds[wave][r][idx]      = (__bf16)p0[v];
      plds[wave][r][idx + 16] = (__bf16)p1[v];
    }
    __syncthreads();

    // preload P and V fragments, then batch the PV WMMAs
    const v16bf pf = load_frag16(&plds[wave][idx][half * 8]);
    v16bf vf[4];
#pragma unroll
    for (int t = 0; t < 4; ++t) vf[t] = load_frag16(&vlds[buf][t * 16 + idx][half * 8]);
#pragma unroll
    for (int t = 0; t < 4; ++t) acc[t] = wmma_bf16(pf, vf[t], acc[t]);
    __syncthreads();                                   // done with buf + plds
  }

#pragma unroll
  for (int t = 0; t < 4; ++t) {
#pragma unroll
    for (int v = 0; v < 8; ++v) {
      const float val = acc[t][v] / (lrow[v] + 1e-8f);
      ctx[((size_t)b * S_ + s0 + v + half * 8) * H_ + h * HD_ + t * 16 + idx] = (__bf16)val;
    }
  }
}

// ---------------- LayerNorm over H per token ----------------
__global__ __launch_bounds__(256) void ln_kernel(
    const float* __restrict__ delta, const float* __restrict__ gamma,
    const float* __restrict__ beta, float* __restrict__ out) {
  __shared__ float red[8];
  const int token = blockIdx.x;
  const int tid = threadIdx.x;
  const int wave = tid >> 5, lane = tid & 31;
  const float* row = delta + (size_t)token * H_;
  float x[4];
  float s = 0.f;
#pragma unroll
  for (int j = 0; j < 4; ++j) { x[j] = row[tid + j * 256]; s += x[j]; }
#pragma unroll
  for (int off = 1; off < 32; off <<= 1) s += __shfl_xor(s, off, 32);
  if (lane == 0) red[wave] = s;
  __syncthreads();
  float tot = 0.f;
#pragma unroll
  for (int w = 0; w < 8; ++w) tot += red[w];
  const float mu = tot * (1.0f / H_);
  float vs = 0.f;
#pragma unroll
  for (int j = 0; j < 4; ++j) { const float d = x[j] - mu; vs += d * d; }
#pragma unroll
  for (int off = 1; off < 32; off <<= 1) vs += __shfl_xor(vs, off, 32);
  __syncthreads();
  if (lane == 0) red[wave] = vs;
  __syncthreads();
  float vtot = 0.f;
#pragma unroll
  for (int w = 0; w < 8; ++w) vtot += red[w];
  const float inv = rsqrtf(vtot * (1.0f / H_) + 1e-5f);
  float* orow = out + (size_t)token * H_;
#pragma unroll
  for (int j = 0; j < 4; ++j) {
    const int col = tid + j * 256;
    orow[col] = (x[j] - mu) * inv * gamma[col] + beta[col];
  }
}

extern "C" void kernel_launch(void* const* d_in, const int* in_sizes, int n_in,
                              void* d_out, int out_size, void* d_ws, size_t ws_size,
                              hipStream_t stream) {
  const float* hs    = (const float*)d_in[0];
  const float* at    = (const float*)d_in[1];
  const int*   msk   = (const int*)d_in[2];
  const float* Wq    = (const float*)d_in[3];
  const float* bq    = (const float*)d_in[4];
  const float* Wk    = (const float*)d_in[5];
  const float* bk    = (const float*)d_in[6];
  const float* Wv    = (const float*)d_in[7];
  const float* bv    = (const float*)d_in[8];
  const float* Wo    = (const float*)d_in[9];
  const float* bo    = (const float*)d_in[10];
  const float* gamma = (const float*)d_in[11];
  const float* beta  = (const float*)d_in[12];
  const float* rs    = (const float*)d_in[13];
  float* out = (float*)d_out;

  const size_t n_hs = (size_t)B_ * S_ * H_;   // 16,777,216
  const size_t n_at = (size_t)B_ * A_ * H_;   //  2,097,152
  const size_t n_w  = (size_t)H_ * H_;        //  1,048,576

  uint8_t* p = (uint8_t*)d_ws;
  __bf16* hs_bf = (__bf16*)p;  p += n_hs * 2;
  __bf16* at_bf = (__bf16*)p;  p += n_at * 2;
  __bf16* wq_bf = (__bf16*)p;  p += n_w * 2;
  __bf16* wk_bf = (__bf16*)p;  p += n_w * 2;
  __bf16* wv_bf = (__bf16*)p;  p += n_w * 2;
  __bf16* wo_bf = (__bf16*)p;  p += n_w * 2;
  __bf16* Qb    = (__bf16*)p;  p += n_hs * 2;
  __bf16* Kb    = (__bf16*)p;  p += n_at * 2;
  __bf16* Vt    = (__bf16*)p;  p += n_at * 2;   // transposed: [H_, B_*A_]
  __bf16* ctxb  = (__bf16*)p;  p += n_hs * 2;
  float*  delta = (float*)p;   p += n_hs * 4;

  const int T = 256;
  cvt_kernel<<<(int)((n_hs + T - 1) / T), T, 0, stream>>>(hs, hs_bf, (int)n_hs, 1);
  cvt_kernel<<<(int)((n_at + T - 1) / T), T, 0, stream>>>(at, at_bf, (int)n_at, 1);
  cvt_kernel<<<(int)((n_w  + T - 1) / T), T, 0, stream>>>(Wq, wq_bf, (int)n_w, 0);
  cvt_kernel<<<(int)((n_w  + T - 1) / T), T, 0, stream>>>(Wk, wk_bf, (int)n_w, 0);
  cvt_kernel<<<(int)((n_w  + T - 1) / T), T, 0, stream>>>(Wv, wv_bf, (int)n_w, 0);
  cvt_kernel<<<(int)((n_w  + T - 1) / T), T, 0, stream>>>(Wo, wo_bf, (int)n_w, 0);

  const int MQ = B_ * S_;   // 16384
  const int MA = B_ * A_;   // 2048
  gemm_kernel<0, H_><<<dim3(MQ / 128, H_ / 64), 256, 0, stream>>>(hs_bf, wq_bf, bq, Qb, nullptr, nullptr, MQ, H_);
  gemm_kernel<0, H_><<<dim3(MA / 128, H_ / 64), 256, 0, stream>>>(at_bf, wk_bf, bk, Kb, nullptr, nullptr, MA, H_);
  gemm_kernel<1, H_><<<dim3(MA / 128, H_ / 64), 256, 0, stream>>>(at_bf, wv_bf, bv, Vt, nullptr, nullptr, MA, H_);

  attn_kernel<<<(B_ * NH_ * (S_ / 16)) / 8, 256, 0, stream>>>(Qb, Kb, Vt, msk, ctxb);

  gemm_kernel<2, H_><<<dim3(MQ / 128, H_ / 64), 256, 0, stream>>>(ctxb, wo_bf, bo, nullptr, delta, rs, MQ, H_);

  ln_kernel<<<MQ, 256, 0, stream>>>(delta, gamma, beta, out);
}
